// TransformerHybridBlock_75007308857548
// MI455X (gfx1250) — compile-verified, matrix operand order checked by reading, not run on hardware
//
#include <hip/hip_runtime.h>

// ---------------------------------------------------------------------------
// Model constants (match reference)
// ---------------------------------------------------------------------------
#define S_LEN   2048
#define DMODEL  1024
#define NHEAD   16
#define NKVH    4
#define HDIM    64
#define LATDIM  256
#define WINLEN  256
#define EPSF    1e-6f

// ---------------------------------------------------------------------------
// CDNA5 WMMA helpers (gfx1250, wave32)
// ---------------------------------------------------------------------------
typedef __bf16        bf16x16 __attribute__((ext_vector_type(16)));
typedef float         f32x8   __attribute__((ext_vector_type(8)));
typedef unsigned int  u32x4   __attribute__((ext_vector_type(4)));

union FragU { bf16x16 bf; u32x4 u[2]; };

__device__ __forceinline__ unsigned short f2bf(float x) {
  unsigned int u = __float_as_uint(x);
  u += 0x7FFFu + ((u >> 16) & 1u);           // round-to-nearest-even
  return (unsigned short)(u >> 16);
}

// Load a 16x32 bf16 fragment (A-layout) from a row-major [rows x stride] array.
// Lane L holds row `row` (caller passes absolute row), elements at K offsets:
//   e<8  -> k0 + (L<16?0:8) + e
//   e>=8 -> k0 + 16 + (L<16?0:8) + (e-8)
// i.e. two contiguous 16-byte loads per lane.
__device__ __forceinline__ bf16x16 load_frag(const unsigned short* base, int stride,
                                             int row, int k0, int lane) {
  const int hi = (lane & 16) ? 8 : 0;
  const unsigned short* p = base + (size_t)row * stride + k0 + hi;
  FragU f;
  f.u[0] = *reinterpret_cast<const u32x4*>(p);
  f.u[1] = *reinterpret_cast<const u32x4*>(p + 16);
  return f.bf;
}

__device__ __forceinline__ f32x8 wmma_bf16(bf16x16 a, bf16x16 b, f32x8 c) {
  return __builtin_amdgcn_wmma_f32_16x16x32_bf16(false, a, false, b, (short)0, c,
                                                 false, false);
}

// ---------------------------------------------------------------------------
// Weight convert: W[f32, K x N] -> Wt[bf16, N x K]  (transpose + downconvert)
// ---------------------------------------------------------------------------
__global__ void thb_wt_kernel(const float* __restrict__ W, unsigned short* __restrict__ Wt,
                              int K, int N) {
  int idx = blockIdx.x * blockDim.x + threadIdx.x;
  if (idx >= K * N) return;
  int n = idx / K, k = idx - n * K;
  Wt[idx] = f2bf(W[(size_t)k * N + n]);
}

// ---------------------------------------------------------------------------
// RMSNorm: xn_bf16[s][c] = bf16( x[s][c] * rsqrt(mean(x^2)+eps) * w[c] )
// one 256-thread block per row, D = 1024
// ---------------------------------------------------------------------------
__global__ __launch_bounds__(256)
void thb_rmsnorm_kernel(const float* __restrict__ x, const float* __restrict__ w,
                        unsigned short* __restrict__ xn) {
  const int s = blockIdx.x;
  const int t = threadIdx.x;
  const float* row = x + (size_t)s * DMODEL;
  float acc = 0.f;
#pragma unroll
  for (int i = 0; i < 4; ++i) {
    float v = row[t + 256 * i];
    acc += v * v;
  }
  __shared__ float red[256];
  red[t] = acc;
  __syncthreads();
  for (int off = 128; off > 0; off >>= 1) {
    if (t < off) red[t] += red[t + off];
    __syncthreads();
  }
  const float rs = rsqrtf(red[0] * (1.0f / DMODEL) + EPSF);
#pragma unroll
  for (int i = 0; i < 4; ++i) {
    int c = t + 256 * i;
    xn[(size_t)s * DMODEL + c] = f2bf(row[c] * rs * w[c]);
  }
}

// ---------------------------------------------------------------------------
// WMMA GEMM: C[M x N] (f32 and/or bf16) = A[bf16, M x K] * Bt[bf16, N x K]^T
// optional residual (f32 [M x N]) added in the epilogue.
// 256 threads = 8 waves; block tile 256(M) x 64(N); wave tile 32 x 64.
// Ping-pong double buffering, K unrolled by 2: two statically distinct
// fragment groups alternate load/consume phases, so there are no register
// copies in the steady state (and no WMMA->VALU WAR-hazard nops on them).
// M % 256 == 0, N % 64 == 0, K % 64 == 0 for all uses here.
// ---------------------------------------------------------------------------
__global__ __launch_bounds__(256)
void thb_gemm_kernel(const unsigned short* __restrict__ A,
                     const unsigned short* __restrict__ Bt,
                     float* Cf, unsigned short* Cbf, const float* resid,
                     int M, int N, int K) {
  const int lane = threadIdx.x & 31;
  const int wave = threadIdx.x >> 5;
  const int row0 = blockIdx.y * 256 + wave * 32;
  const int col0 = blockIdx.x * 64;
  const int nc = lane & 15;

  f32x8 acc[2][4];
#pragma unroll
  for (int mi = 0; mi < 2; ++mi)
#pragma unroll
    for (int j = 0; j < 4; ++j)
      acc[mi][j] = (f32x8){0.f,0.f,0.f,0.f,0.f,0.f,0.f,0.f};

  const int ar0 = row0 + nc;
  const int ar1 = row0 + 16 + nc;
  const int bc0 = col0 + nc;

  bf16x16 a0A, a1A, bA[4];     // ping buffer
  bf16x16 a0B, a1B, bB[4];     // pong buffer

  auto loadA = [&](int k0) {
    a0A = load_frag(A, K, ar0, k0, lane);
    a1A = load_frag(A, K, ar1, k0, lane);
#pragma unroll
    for (int j = 0; j < 4; ++j) bA[j] = load_frag(Bt, K, bc0 + j * 16, k0, lane);
  };
  auto loadB = [&](int k0) {
    a0B = load_frag(A, K, ar0, k0, lane);
    a1B = load_frag(A, K, ar1, k0, lane);
#pragma unroll
    for (int j = 0; j < 4; ++j) bB[j] = load_frag(Bt, K, bc0 + j * 16, k0, lane);
  };
  auto computeA = [&]() {
#pragma unroll
    for (int j = 0; j < 4; ++j) {
      acc[0][j] = wmma_bf16(a0A, bA[j], acc[0][j]);
      acc[1][j] = wmma_bf16(a1A, bA[j], acc[1][j]);
    }
  };
  auto computeB = [&]() {
#pragma unroll
    for (int j = 0; j < 4; ++j) {
      acc[0][j] = wmma_bf16(a0B, bB[j], acc[0][j]);
      acc[1][j] = wmma_bf16(a1B, bB[j], acc[1][j]);
    }
  };

  // prologue
  loadA(0);
  int k0 = 0;
  // steady state: two 32-wide K-steps per iteration, zero register rotation
  for (; k0 + 64 < K; k0 += 64) {
    loadB(k0 + 32);
    computeA();
    loadA(k0 + 64);
    computeB();
  }
  // tail: k0 == K - 64
  loadB(k0 + 32);
  computeA();
  computeB();

  const int hi = (lane & 16) ? 8 : 0;
#pragma unroll
  for (int mi = 0; mi < 2; ++mi) {
#pragma unroll
    for (int j = 0; j < 4; ++j) {
#pragma unroll
      for (int r = 0; r < 8; ++r) {
        const int row = row0 + mi * 16 + hi + r;
        const int col = col0 + j * 16 + nc;
        const size_t idx = (size_t)row * N + col;
        float v = acc[mi][j][r];
        if (resid) v += resid[idx];
        if (Cf)  Cf[idx]  = v;
        if (Cbf) Cbf[idx] = f2bf(v);
      }
    }
  }
}

// ---------------------------------------------------------------------------
// RoPE + qk-norm + head-major relayout.
// x[f32, S x (nheads*64)] -> dst[bf16, nheads x S x 64]
// one 32-lane wave per (s, h); lane handles one even/odd pair.
// ---------------------------------------------------------------------------
__global__ __launch_bounds__(32)
void thb_rope_qknorm_kernel(const float* __restrict__ x, unsigned short* __restrict__ dst,
                            int nheads, int S, float base) {
  const int s = blockIdx.x;
  const int h = blockIdx.y;
  const int p = threadIdx.x;          // pair index 0..31
  const float* src = x + ((size_t)s * nheads + h) * HDIM;
  const float xe = src[2 * p];
  const float xo = src[2 * p + 1];
  const float invf = __powf(base, -((float)(2 * p) * (1.0f / HDIM)));
  const float ang = (float)s * invf;
  float si, co;
  __sincosf(ang, &si, &co);
  const float re = xe * co - xo * si;
  const float ro = xe * si + xo * co;
  float sum = re * re + ro * ro;
#pragma unroll
  for (int off = 1; off < 32; off <<= 1) sum += __shfl_xor(sum, off, 32);
  const float rs = rsqrtf(sum * (1.0f / HDIM) + EPSF);
  unsigned short* d = dst + ((size_t)h * S + s) * HDIM;
  d[2 * p]     = f2bf(re * rs);
  d[2 * p + 1] = f2bf(ro * rs);
}

// ---------------------------------------------------------------------------
// V relayout: v[f32, S x (nkvh*64)] -> Vt[bf16, nkvh x 64 x S] (transposed)
// ---------------------------------------------------------------------------
__global__ __launch_bounds__(64)
void thb_vt_kernel(const float* __restrict__ v, unsigned short* __restrict__ Vt,
                   int nheads, int S) {
  const int s = blockIdx.x, h = blockIdx.y, d = threadIdx.x;
  Vt[((size_t)h * HDIM + d) * S + s] = f2bf(v[((size_t)s * nheads + h) * HDIM + d]);
}

// ---------------------------------------------------------------------------
// Flash attention (wave32 WMMA). One wave per (16-query block, head).
// Qh[bf16, H x S x 64], Kh[bf16, KVH x S x 64], Vt[bf16, KVH x 64 x S].
// win == 0 -> pure causal; win > 0 -> causal sliding window of `win`.
// Output: Obf[bf16, S x D] (heads merged).
// ---------------------------------------------------------------------------
__global__ __launch_bounds__(32)
void thb_flash_kernel(const unsigned short* __restrict__ Qh,
                      const unsigned short* __restrict__ Kh,
                      const unsigned short* __restrict__ Vt,
                      unsigned short* __restrict__ Obf, int S, int win) {
  __shared__ __align__(16) unsigned short plds[16 * 32];

  const int lane = threadIdx.x;
  const int q0 = blockIdx.x * 16;
  const int h  = blockIdx.y;
  const int kvh = h >> 2;                   // H/KVH == 4 (repeat_kv)
  const int nc = lane & 15;
  const int hi = (lane & 16) ? 8 : 0;

  const unsigned short* Qb = Qh + (size_t)h   * S * HDIM;
  const unsigned short* Kb = Kh + (size_t)kvh * S * HDIM;
  const unsigned short* Vb = Vt + (size_t)kvh * HDIM * S;

  const bf16x16 qf0 = load_frag(Qb, HDIM, q0 + nc, 0, lane);
  const bf16x16 qf1 = load_frag(Qb, HDIM, q0 + nc, 32, lane);

  f32x8 o[4];
#pragma unroll
  for (int j = 0; j < 4; ++j) o[j] = (f32x8){0.f,0.f,0.f,0.f,0.f,0.f,0.f,0.f};
  float m[8], l[8];
#pragma unroll
  for (int r = 0; r < 8; ++r) { m[r] = -1e30f; l[r] = 0.f; }

  int kb_start = 0;
  if (win > 0) {
    int st = q0 - (win - 1);
    if (st < 0) st = 0;
    kb_start = st & ~31;
  }

  const f32x8 zero8 = (f32x8){0.f,0.f,0.f,0.f,0.f,0.f,0.f,0.f};

  for (int kb = kb_start; kb <= q0 + 15; kb += 32) {
    // ---- scores: two 16x16 tiles, K-dim = 64 (2 WMMAs each) ----
    const int key0 = kb + nc;
    const int key1 = kb + 16 + nc;
    const int kr0 = key0 < S ? key0 : S - 1;
    const int kr1 = key1 < S ? key1 : S - 1;

    bf16x16 kf;
    kf = load_frag(Kb, HDIM, kr0, 0, lane);
    f32x8 s0 = wmma_bf16(qf0, kf, zero8);
    kf = load_frag(Kb, HDIM, kr0, 32, lane);
    s0 = wmma_bf16(qf1, kf, s0);
    kf = load_frag(Kb, HDIM, kr1, 0, lane);
    f32x8 s1 = wmma_bf16(qf0, kf, zero8);
    kf = load_frag(Kb, HDIM, kr1, 32, lane);
    s1 = wmma_bf16(qf1, kf, s1);

    // ---- hoist V fragment loads: independent of softmax, overlap with VALU ----
    bf16x16 vfr[4];
#pragma unroll
    for (int j = 0; j < 4; ++j) vfr[j] = load_frag(Vb, S, j * 16 + nc, kb, lane);

    // ---- mask + online softmax (per-lane rows r + hi) ----
#pragma unroll
    for (int r = 0; r < 8; ++r) {
      const int q = q0 + hi + r;
      const bool v0 = (key0 <= q) && (win == 0 || (q - key0) < win);
      const bool v1 = (key1 <= q) && (win == 0 || (q - key1) < win);
      const float a0 = v0 ? s0[r] * 0.125f : -1e30f;   // 1/sqrt(64)
      const float a1 = v1 ? s1[r] * 0.125f : -1e30f;
      float rm = fmaxf(a0, a1);
#pragma unroll
      for (int off = 1; off < 16; off <<= 1) rm = fmaxf(rm, __shfl_xor(rm, off, 32));
      const float mn = fmaxf(m[r], rm);
      const float sc = __expf(m[r] - mn);
      const float p0 = v0 ? __expf(a0 - mn) : 0.f;
      const float p1 = v1 ? __expf(a1 - mn) : 0.f;
      float rs = p0 + p1;
#pragma unroll
      for (int off = 1; off < 16; off <<= 1) rs += __shfl_xor(rs, off, 32);
      l[r] = l[r] * sc + rs;
      m[r] = mn;
#pragma unroll
      for (int j = 0; j < 4; ++j) o[j][r] *= sc;
      plds[(hi + r) * 32 + nc]      = f2bf(p0);
      plds[(hi + r) * 32 + 16 + nc] = f2bf(p1);
    }
    asm volatile("s_wait_dscnt 0x0" ::: "memory");

    // ---- re-layout P (C-frag -> A-frag) via LDS ----
    FragU pf;
    {
      const unsigned short* pp = plds + nc * 32 + hi; // hi==8 for lanes 16..31
      pf.u[0] = *reinterpret_cast<const u32x4*>(pp);
      pf.u[1] = *reinterpret_cast<const u32x4*>(pp + 16);
    }

    // ---- PV: O[16 x 64] += P[16 x 32] * V[32 x 64] (Vt rows = dims) ----
#pragma unroll
    for (int j = 0; j < 4; ++j) {
      o[j] = wmma_bf16(pf.bf, vfr[j], o[j]);
    }
  }

  // ---- epilogue: normalize + merge heads ----
#pragma unroll
  for (int r = 0; r < 8; ++r) {
    const float inv = (l[r] > 0.f) ? (1.f / l[r]) : 0.f;
    const int row = q0 + hi + r;
#pragma unroll
    for (int j = 0; j < 4; ++j) {
      Obf[(size_t)row * DMODEL + h * HDIM + j * 16 + nc] = f2bf(o[j][r] * inv);
    }
  }
}

// ---------------------------------------------------------------------------
// Host-side orchestration
// ---------------------------------------------------------------------------
extern "C" void kernel_launch(void* const* d_in, const int* in_sizes, int n_in,
                              void* d_out, int out_size, void* d_ws, size_t ws_size,
                              hipStream_t stream) {
  (void)in_sizes; (void)n_in; (void)out_size; (void)ws_size;

  const float* hidden  = (const float*)d_in[0];
  const float* norm1_w = (const float*)d_in[1];
  const float* norm2_w = (const float*)d_in[2];
  const float* g_wq    = (const float*)d_in[3];
  const float* g_kvd   = (const float*)d_in[4];
  const float* g_kup   = (const float*)d_in[5];
  const float* g_vup   = (const float*)d_in[6];
  const float* g_wo    = (const float*)d_in[7];
  const float* l_wq    = (const float*)d_in[8];
  const float* l_wk    = (const float*)d_in[9];
  const float* l_wv    = (const float*)d_in[10];
  const float* l_wo    = (const float*)d_in[11];
  float* out = (float*)d_out;

  // ---- workspace carve (256B aligned) ----
  char* p = (char*)d_ws;
  auto alloc = [&](size_t bytes) -> void* {
    void* r = (void*)p;
    p += (bytes + 255) & ~(size_t)255;
    return r;
  };
  unsigned short* xn    = (unsigned short*)alloc((size_t)S_LEN * DMODEL * 2);
  float*          qf    = (float*)         alloc((size_t)S_LEN * DMODEL * 4);
  float*          kf    = (float*)         alloc((size_t)S_LEN * LATDIM * 4);
  float*          vf    = (float*)         alloc((size_t)S_LEN * LATDIM * 4);
  unsigned short* latbf = (unsigned short*)alloc((size_t)S_LEN * LATDIM * 2);
  unsigned short* Qh    = (unsigned short*)alloc((size_t)NHEAD * S_LEN * HDIM * 2);
  unsigned short* Kh    = (unsigned short*)alloc((size_t)NKVH * S_LEN * HDIM * 2);
  unsigned short* Vt    = (unsigned short*)alloc((size_t)NKVH * HDIM * S_LEN * 2 + 256);
  unsigned short* attn  = (unsigned short*)alloc((size_t)S_LEN * DMODEL * 2);
  float*          xmid  = (float*)         alloc((size_t)S_LEN * DMODEL * 4);
  unsigned short* wqT   = (unsigned short*)alloc((size_t)DMODEL * DMODEL * 2);
  unsigned short* woT   = (unsigned short*)alloc((size_t)DMODEL * DMODEL * 2);
  unsigned short* wkT   = (unsigned short*)alloc((size_t)LATDIM * DMODEL * 2);
  unsigned short* wvT   = (unsigned short*)alloc((size_t)LATDIM * DMODEL * 2);
  unsigned short* kupT  = (unsigned short*)alloc((size_t)LATDIM * LATDIM * 2);
  unsigned short* vupT  = (unsigned short*)alloc((size_t)LATDIM * LATDIM * 2);

  const dim3 blk256(256), blk32(32), blk64(64);
  auto wt_grid = [](int K, int N) { return dim3((K * N + 255) / 256); };
  const dim3 gemm_blk(256);
  const dim3 rope_grid_q(S_LEN, NHEAD), rope_grid_k(S_LEN, NKVH);
  const dim3 vt_grid(S_LEN, NKVH);
  const dim3 flash_grid(S_LEN / 16, NHEAD);

  // ========================= Sublayer 1: global latent =====================
  thb_wt_kernel<<<wt_grid(DMODEL, DMODEL), blk256, 0, stream>>>(g_wq,  wqT,  DMODEL, DMODEL);
  thb_wt_kernel<<<wt_grid(DMODEL, LATDIM), blk256, 0, stream>>>(g_kvd, wkT,  DMODEL, LATDIM);
  thb_wt_kernel<<<wt_grid(LATDIM, LATDIM), blk256, 0, stream>>>(g_kup, kupT, LATDIM, LATDIM);
  thb_wt_kernel<<<wt_grid(LATDIM, LATDIM), blk256, 0, stream>>>(g_vup, vupT, LATDIM, LATDIM);
  thb_wt_kernel<<<wt_grid(DMODEL, DMODEL), blk256, 0, stream>>>(g_wo,  woT,  DMODEL, DMODEL);

  thb_rmsnorm_kernel<<<S_LEN, blk256, 0, stream>>>(hidden, norm1_w, xn);

  thb_gemm_kernel<<<dim3(DMODEL / 64, S_LEN / 256), gemm_blk, 0, stream>>>(
      xn, wqT, qf, nullptr, nullptr, S_LEN, DMODEL, DMODEL);
  thb_gemm_kernel<<<dim3(LATDIM / 64, S_LEN / 256), gemm_blk, 0, stream>>>(
      xn, wkT, nullptr, latbf, nullptr, S_LEN, LATDIM, DMODEL);
  thb_gemm_kernel<<<dim3(LATDIM / 64, S_LEN / 256), gemm_blk, 0, stream>>>(
      latbf, kupT, kf, nullptr, nullptr, S_LEN, LATDIM, LATDIM);
  thb_gemm_kernel<<<dim3(LATDIM / 64, S_LEN / 256), gemm_blk, 0, stream>>>(
      latbf, vupT, vf, nullptr, nullptr, S_LEN, LATDIM, LATDIM);

  thb_rope_qknorm_kernel<<<rope_grid_q, blk32, 0, stream>>>(qf, Qh, NHEAD, S_LEN, 1000000.0f);
  thb_rope_qknorm_kernel<<<rope_grid_k, blk32, 0, stream>>>(kf, Kh, NKVH, S_LEN, 1000000.0f);
  thb_vt_kernel<<<vt_grid, blk64, 0, stream>>>(vf, Vt, NKVH, S_LEN);

  thb_flash_kernel<<<flash_grid, blk32, 0, stream>>>(Qh, Kh, Vt, attn, S_LEN, 0);

  thb_gemm_kernel<<<dim3(DMODEL / 64, S_LEN / 256), gemm_blk, 0, stream>>>(
      attn, woT, xmid, nullptr, hidden, S_LEN, DMODEL, DMODEL);

  // ========================= Sublayer 2: local GQA =========================
  thb_wt_kernel<<<wt_grid(DMODEL, DMODEL), blk256, 0, stream>>>(l_wq, wqT, DMODEL, DMODEL);
  thb_wt_kernel<<<wt_grid(DMODEL, LATDIM), blk256, 0, stream>>>(l_wk, wkT, DMODEL, LATDIM);
  thb_wt_kernel<<<wt_grid(DMODEL, LATDIM), blk256, 0, stream>>>(l_wv, wvT, DMODEL, LATDIM);
  thb_wt_kernel<<<wt_grid(DMODEL, DMODEL), blk256, 0, stream>>>(l_wo, woT, DMODEL, DMODEL);

  thb_rmsnorm_kernel<<<S_LEN, blk256, 0, stream>>>(xmid, norm2_w, xn);

  thb_gemm_kernel<<<dim3(DMODEL / 64, S_LEN / 256), gemm_blk, 0, stream>>>(
      xn, wqT, qf, nullptr, nullptr, S_LEN, DMODEL, DMODEL);
  thb_gemm_kernel<<<dim3(LATDIM / 64, S_LEN / 256), gemm_blk, 0, stream>>>(
      xn, wkT, kf, nullptr, nullptr, S_LEN, LATDIM, DMODEL);
  thb_gemm_kernel<<<dim3(LATDIM / 64, S_LEN / 256), gemm_blk, 0, stream>>>(
      xn, wvT, vf, nullptr, nullptr, S_LEN, LATDIM, DMODEL);

  thb_rope_qknorm_kernel<<<rope_grid_q, blk32, 0, stream>>>(qf, Qh, NHEAD, S_LEN, 10000.0f);
  thb_rope_qknorm_kernel<<<rope_grid_k, blk32, 0, stream>>>(kf, Kh, NKVH, S_LEN, 10000.0f);
  thb_vt_kernel<<<vt_grid, blk64, 0, stream>>>(vf, Vt, NKVH, S_LEN);

  thb_flash_kernel<<<flash_grid, blk32, 0, stream>>>(Qh, Kh, Vt, attn, S_LEN, WINLEN);

  thb_gemm_kernel<<<dim3(DMODEL / 64, S_LEN / 256), gemm_blk, 0, stream>>>(
      attn, woT, out, nullptr, xmid, S_LEN, DMODEL, DMODEL);
}
